// Attention_org_35261681500333
// MI455X (gfx1250) — matrix operand order repaired and checked by reading, not compiled, not run on hardware
//
#include <hip/hip_runtime.h>
#include <math.h>

// ---------------------------------------------------------------------------
// Multi-scale channel attention for MI455X (gfx1250, wave32, WMMA).
// v3: every hot GEMM loop reads unguarded, per-lane-contiguous f16:
//  - heads regrouping materialized once (divergent channel map OUT of loops)
//  - weights pre-converted to f16 in B-fragment-friendly [n][k] layout
//  - Qh/Kh stored transposed (channel-major) so the K=1024 contraction is
//    b128 loads on both operands, no guards, no cvt.
// All GEMMs: v_wmma_f32_16x16x32_f16 (f16 A/B, f32 accumulate).
// ---------------------------------------------------------------------------

typedef __attribute__((ext_vector_type(16))) _Float16 v16h;
typedef __attribute__((ext_vector_type(8)))  float    v8f;

#define BQ    16      // batch
#define NSEQ  1024    // sequence length
#define NH    4       // heads
#define KVC   240     // per-head K/V channels (960/4)
#define KVP   256     // padded channel dim for f16 intermediates
#define WPB   8       // waves per block (256 threads)

__device__ __forceinline__ int lane_id() { return (int)(threadIdx.x & 31); }

// ---- WMMA fragment loaders (ISA 7.12.2 layouts) -----------------------------
// A (16x32, MxK): lane L: M=L&15, hi=L>>4; V0..3: K=hi*8+2v(+1); V4..7:
// K=16+hi*8+2(v-4)(+1).  element(m,k) = p[m*sm + k*sk].
// With sk==1 each lane reads two contiguous 16B runs -> global_load_b128.
__device__ __forceinline__ v16h load_a_f16(const _Float16* p, long sm, long sk,
                                           int k0) {
  const int lane = lane_id(), hi = lane >> 4, m = lane & 15;
  v16h a;
#pragma unroll
  for (int v = 0; v < 8; ++v) {
    const int kb = (v < 4) ? (hi * 8 + 2 * v) : (16 + hi * 8 + 2 * (v - 4));
    const long base = (long)m * sm + (long)(k0 + kb) * sk;
    a[2 * v]     = p[base];
    a[2 * v + 1] = p[base + sk];
  }
  return a;
}

// B (32x16, KxN): lane L: N=L&15, hi=L>>4; V0..3: K=hi*8+2v(+1); V4..7:
// K=16+hi*8+2(v-4)(+1) (same as A): with sk==1 two 16B runs -> b128.
// element(k,n) = p[k*sk + n*sn].
__device__ __forceinline__ v16h load_b_f16(const _Float16* p, long sk, long sn,
                                           int k0) {
  const int lane = lane_id(), hi = lane >> 4, n = lane & 15;
  v16h b;
#pragma unroll
  for (int v = 0; v < 8; ++v) {
    const int kb = (v < 4) ? (hi * 8 + 2 * v) : (16 + hi * 8 + 2 * (v - 4));
    const long base = (long)(k0 + kb) * sk + (long)n * sn;
    b[2 * v]     = p[base];
    b[2 * v + 1] = p[base + sk];
  }
  return b;
}

// f32-source A loader, K-guarded (only used in the tiny Q projection)
__device__ __forceinline__ v16h load_a_f32g(const float* p, long sm, int k0,
                                            int klim) {
  const int lane = lane_id(), hi = lane >> 4, m = lane & 15;
  v16h a;
#pragma unroll
  for (int v = 0; v < 8; ++v) {
    const int kb = (v < 4) ? (hi * 8 + 2 * v) : (16 + hi * 8 + 2 * (v - 4));
    const int k = k0 + kb;
    a[2 * v]     = (k     < klim) ? (_Float16)p[(long)m * sm + k]     : (_Float16)0.f;
    a[2 * v + 1] = (k + 1 < klim) ? (_Float16)p[(long)m * sm + k + 1] : (_Float16)0.f;
  }
  return a;
}

// C/D (16x16 f32): lane L: N=L&15; VGPR v: M = v + 8*(L>>4).
__device__ __forceinline__ void store_d_f32(float* p, long sm, long sn,
                                            const v8f& c, float scale) {
  const int lane = lane_id(), hi = lane >> 4, n = lane & 15;
#pragma unroll
  for (int v = 0; v < 8; ++v)
    p[(long)(v + 8 * hi) * sm + (long)n * sn] = c[v] * scale;
}

__device__ __forceinline__ void store_d_f16(_Float16* p, long sm, long sn,
                                            const v8f& c) {
  const int lane = lane_id(), hi = lane >> 4, n = lane & 15;
#pragma unroll
  for (int v = 0; v < 8; ++v)
    p[(long)(v + 8 * hi) * sm + (long)n * sn] = (_Float16)c[v];
}

__device__ __forceinline__ v8f wmma_f16(v16h a, v16h b, v8f c) {
  // (neg_a, A, neg_b, B, c_mod, C, reuse_a, reuse_b)
  return __builtin_amdgcn_wmma_f32_16x16x32_f16(false, a, false, b, (short)0, c,
                                                false, false);
}

// ---------------------------------------------------------------------------
// Prep kernels (run once per call, tiny): move all divergence/conversion here.
// ---------------------------------------------------------------------------

// emb_all[...,960] -> heads_f16 [B,H,N,256] (j>=240 zero-padded)
__device__ __forceinline__ int head_ch(int h, int j) {
  if (j < 16)  return h * 16 + j;
  if (j < 48)  return 64  + h * 32 + (j - 16);
  if (j < 112) return 192 + h * 64 + (j - 48);
  return 448 + h * 128 + (j - 112);
}

__global__ __launch_bounds__(256)
void regroup_kernel(const float* __restrict__ emb_all,
                    _Float16* __restrict__ heads) {
  const long i = (long)blockIdx.x * 256 + threadIdx.x;   // B*H*N*256
  const int j  = (int)(i & (KVP - 1));
  const long r = i >> 8;                 // (bh, n) row
  const int n  = (int)(r & (NSEQ - 1));
  const int bh = (int)(r >> 10);
  const int h = bh & (NH - 1), b = bh >> 2;
  float x = 0.f;
  if (j < KVC) x = emb_all[((long)b * NSEQ + n) * 960 + head_ch(h, j)];
  heads[i] = (_Float16)x;
}

// Wk/Wv [240,240] f32 -> [240,256] f16 (k padded with zeros)
__global__ __launch_bounds__(256)
void wkv_prep_kernel(const float* __restrict__ Wk, const float* __restrict__ Wv,
                     _Float16* __restrict__ Wkf, _Float16* __restrict__ Wvf) {
  const int i = blockIdx.x * 256 + threadIdx.x;   // 240*256
  if (i >= KVC * KVP) return;
  const int k = i & (KVP - 1), e = i >> 8;
  Wkf[i] = (k < KVC) ? (_Float16)Wk[e * KVC + k] : (_Float16)0.f;
  Wvf[i] = (k < KVC) ? (_Float16)Wv[e * KVC + k] : (_Float16)0.f;
}

// Wq [H,d,d] f32 -> [H,d,dp] f16 (k padded to dp = max(d,32))
__global__ __launch_bounds__(256)
void wq_prep_kernel(const float* __restrict__ Wq, _Float16* __restrict__ Wqf,
                    int d, int dp) {
  const int i = blockIdx.x * 256 + threadIdx.x;   // H*d*dp
  if (i >= NH * d * dp) return;
  const int k = i % dp, e = (i / dp) % d, h = i / (dp * d);
  Wqf[i] = (k < d) ? (_Float16)Wq[(h * d + e) * d + k] : (_Float16)0.f;
}

// Wo [C,C] f32 -> f16 (no pad needed; C % 32 == 0)
__global__ __launch_bounds__(256)
void wo_prep_kernel(const float* __restrict__ Wo, _Float16* __restrict__ Wof,
                    int n) {
  const int i = blockIdx.x * 256 + threadIdx.x;
  if (i < n) Wof[i] = (_Float16)Wo[i];
}

// zero the 16 pad columns (240..255) of Vh
__global__ __launch_bounds__(256)
void padv_kernel(_Float16* __restrict__ Vh) {
  const long rows = (long)BQ * NH * NSEQ;   // 65536
  const long i = (long)blockIdx.x * 256 + threadIdx.x;
  if (i >= rows) return;
  const float4 z = {0.f, 0.f, 0.f, 0.f};
  float4* v = (float4*)(Vh + i * KVP + KVC); v[0] = z; v[1] = z;
}

// ---------------------------------------------------------------------------
// Kernel 1: KhT[b,h,e,n] = sum_d heads[n,d]*Wk[e,d];  Vh[b,h,n,e] likewise.
// K stored channel-major for the scores GEMM; one A feeds 6 wmma per step.
// ---------------------------------------------------------------------------
__global__ __launch_bounds__(256)
void kv_gemm_kernel(const _Float16* __restrict__ heads,
                    const _Float16* __restrict__ Wkf,
                    const _Float16* __restrict__ Wvf,
                    _Float16* __restrict__ KhT, _Float16* __restrict__ Vh) {
  const int wt = blockIdx.x * WPB + (threadIdx.x >> 5);
  const int total = BQ * NH * (NSEQ / 16) * 5;   // 15 e-tiles in blocks of 3
  if (wt >= total) return;
  const int nb = wt % 5;
  const int mt = (wt / 5) % (NSEQ / 16);
  const int bh = wt / (5 * (NSEQ / 16));
  const int e0 = nb * 48;

  const _Float16* A = heads + ((long)bh * NSEQ + mt * 16) * KVP;  // sm=256, sk=1
  v8f ck[3] = {}; v8f cv[3] = {};
  for (int k0 = 0; k0 < KVP; k0 += 32) {   // 8 clean steps (zero-padded)
    const v16h a = load_a_f16(A, KVP, 1, k0);
#pragma unroll
    for (int j = 0; j < 3; ++j) {
      // elem(k,n) = W[(e0+j*16+n)*256 + k]  (sk=1: contiguous per lane)
      const v16h bk = load_b_f16(Wkf + (long)(e0 + j * 16) * KVP, 1, KVP, k0);
      ck[j] = wmma_f16(a, bk, ck[j]);
      const v16h bv = load_b_f16(Wvf + (long)(e0 + j * 16) * KVP, 1, KVP, k0);
      cv[j] = wmma_f16(a, bv, cv[j]);
    }
  }
#pragma unroll
  for (int j = 0; j < 3; ++j) {
    // KhT[bh][e][n]: elem(m=n_row, n=e) -> sm=1, sn=1024 (contiguous per lane)
    store_d_f16(KhT + ((long)bh * KVC + e0 + j * 16) * NSEQ + mt * 16, 1, NSEQ, ck[j]);
    // Vh[bh][n][e]
    store_d_f16(Vh + ((long)bh * NSEQ + mt * 16) * KVP + e0 + j * 16, KVP, 1, cv[j]);
  }
}

// ---------------------------------------------------------------------------
// Kernel 2: QhT[b,h,qoff+e,n] = sum_d emb[b,n,h*d+dd] * Wq[h,e,dd]
// ---------------------------------------------------------------------------
__global__ __launch_bounds__(256)
void q_gemm_kernel(const float* __restrict__ emb, const _Float16* __restrict__ Wqf,
                   _Float16* __restrict__ QhT, int d, int dp, int qoff, int C) {
  const int dt = d / 16;
  const int wt = blockIdx.x * WPB + (threadIdx.x >> 5);
  const int total = BQ * NH * (NSEQ / 16) * dt;
  if (wt >= total) return;
  const int et = wt % dt;
  const int mt = (wt / dt) % (NSEQ / 16);
  const int bh = wt / (dt * (NSEQ / 16));
  const int b = bh / NH, h = bh % NH;

  const float* A = emb + ((long)b * NSEQ + mt * 16) * C + h * d;  // sm=C, sk=1
  const _Float16* Bm = Wqf + (long)h * d * dp + (long)(et * 16) * dp;
  v8f c = {};
  for (int k0 = 0; k0 < dp; k0 += 32) {     // B zero-padded; A guarded (f32)
    const v16h a  = load_a_f32g(A, C, k0, d);
    const v16h bb = load_b_f16(Bm, 1, dp, k0);
    c = wmma_f16(a, bb, c);
  }
  // QhT[bh][qoff+e][n]: sm=1, sn=1024
  store_d_f16(QhT + ((long)bh * KVC + qoff + et * 16) * NSEQ + mt * 16, 1, NSEQ, c);
}

// ---------------------------------------------------------------------------
// Kernel 3: scores[dd,kk] = (1/sqrt(960)) * sum_n QhT[dd,n] * KhT[kk,n]
// K = 1024; both operands channel-major -> per-lane-contiguous b128 loads.
// ---------------------------------------------------------------------------
__global__ __launch_bounds__(256)
void scores_gemm_kernel(const _Float16* __restrict__ QhT,
                        const _Float16* __restrict__ KhT,
                        float* __restrict__ scores, int d, int qoff) {
  const int dt = d / 16;
  const int wt = blockIdx.x * WPB + (threadIdx.x >> 5);
  const int total = BQ * NH * dt * 5;
  if (wt >= total) return;
  const int nb = wt % 5;
  const int mt = (wt / 5) % dt;
  const int bh = wt / (5 * dt);
  const int kk0 = nb * 48;

  const _Float16* A  = QhT + ((long)bh * KVC + qoff + mt * 16) * NSEQ; // sm=1024, sk=1
  const _Float16* Bm = KhT + ((long)bh * KVC + kk0) * NSEQ;            // elem(k,n)=Bm[n*1024+k]
  v8f c[3] = {};
  for (int k0 = 0; k0 < NSEQ; k0 += 32) {
    if (k0 + 128 < NSEQ) {               // streaming prefetch -> global_prefetch_b8
      __builtin_prefetch(A  + k0 + 128, 0, 1);
      __builtin_prefetch(Bm + k0 + 128, 0, 1);
    }
    const v16h a = load_a_f16(A, NSEQ, 1, k0);
#pragma unroll
    for (int j = 0; j < 3; ++j) {
      const v16h bb = load_b_f16(Bm + (long)(j * 16) * NSEQ, 1, NSEQ, k0);
      c[j] = wmma_f16(a, bb, c[j]);
    }
  }
  float* o = scores + ((long)bh * KVC + qoff + mt * 16) * KVC + kk0;
#pragma unroll
  for (int j = 0; j < 3; ++j)
    store_d_f32(o + j * 16, KVC, 1, c[j], 0.0322748612183951f /* 1/sqrt(960) */);
}

// ---------------------------------------------------------------------------
// Kernel 4: instance-norm over (d x 240) per (b,h), row softmax, emit f16
// probs padded to 256 cols (pad = 0 so the K=256 ctx contraction is exact).
// ---------------------------------------------------------------------------
__global__ __launch_bounds__(256)
void norm_softmax_kernel(const float* __restrict__ scores,
                         _Float16* __restrict__ probs, int d, int qoff) {
  const int bh = blockIdx.x;
  const float* S = scores + ((long)bh * KVC + qoff) * KVC;   // d x 240 contiguous
  _Float16* P = probs + ((long)bh * KVC + qoff) * KVP;       // d x 256
  const int n = d * KVC;

  __shared__ float r1[256], r2[256];
  float s1 = 0.f, s2 = 0.f;
  for (int i = threadIdx.x; i < n; i += 256) {
    const float x = S[i];
    s1 += x; s2 += x * x;
  }
  r1[threadIdx.x] = s1; r2[threadIdx.x] = s2;
  __syncthreads();
  for (int off = 128; off > 0; off >>= 1) {
    if ((int)threadIdx.x < off) {
      r1[threadIdx.x] += r1[threadIdx.x + off];
      r2[threadIdx.x] += r2[threadIdx.x + off];
    }
    __syncthreads();
  }
  const float mean = r1[0] / (float)n;
  const float var  = r2[0] / (float)n - mean * mean;   // biased var (jnp.var)
  const float inv  = rsqrtf(var + 1e-5f);

  const int wid = threadIdx.x >> 5, lane = threadIdx.x & 31;
  for (int r = wid; r < d; r += WPB) {
    const float* row = S + (long)r * KVC;
    float mx = -3.4e38f;
    for (int c = lane; c < KVC; c += 32) mx = fmaxf(mx, (row[c] - mean) * inv);
    for (int off = 16; off > 0; off >>= 1) mx = fmaxf(mx, __shfl_xor(mx, off, 32));
    float sum = 0.f;
    for (int c = lane; c < KVC; c += 32) sum += __expf((row[c] - mean) * inv - mx);
    for (int off = 16; off > 0; off >>= 1) sum += __shfl_xor(sum, off, 32);
    const float rs = 1.f / sum;
    for (int c = lane; c < KVP; c += 32) {
      const float e = (c < KVC) ? __expf((row[c] - mean) * inv - mx) * rs : 0.f;
      P[(long)r * KVP + c] = (_Float16)e;
    }
  }
}

// ---------------------------------------------------------------------------
// Kernel 5: ctx[dd,n] = sum_k probs[dd,k] * Vh[n,k]  (K=256, zero-padded)
// Stored pre-flattened as ctxFlat[b, n, dd*H + h] (f16); A feeds 4 N-tiles.
// ---------------------------------------------------------------------------
__global__ __launch_bounds__(256)
void ctx_gemm_kernel(const _Float16* __restrict__ probs,
                     const _Float16* __restrict__ Vh,
                     _Float16* __restrict__ ctx, int d, int qoff, int C) {
  const int dt = d / 16;
  const int wt = blockIdx.x * WPB + (threadIdx.x >> 5);
  const int total = BQ * NH * dt * 16;   // 64 n-tiles in blocks of 4
  if (wt >= total) return;
  const int nb = wt % 16;
  const int mt = (wt / 16) % dt;
  const int bh = wt / (16 * dt);
  const int b = bh / NH, h = bh % NH;
  const int n0 = nb * 64;

  const _Float16* A = probs + ((long)bh * KVC + qoff + mt * 16) * KVP; // sm=256, sk=1
  v8f c[4] = {};
  for (int k0 = 0; k0 < KVP; k0 += 32) {
    const v16h a = load_a_f16(A, KVP, 1, k0);
#pragma unroll
    for (int j = 0; j < 4; ++j) {
      // elem(k,n) = Vh[(n0+j*16+n)*256 + k]  (sk=1: contiguous per lane)
      const v16h bb = load_b_f16(Vh + ((long)bh * NSEQ + n0 + j * 16) * KVP,
                                 1, KVP, k0);
      c[j] = wmma_f16(a, bb, c[j]);
    }
  }
#pragma unroll
  for (int j = 0; j < 4; ++j)
    // elem(m,n) -> ctx[(b*N + n0+j*16 + n)*C + (mt*16 + m)*H + h]
    store_d_f16(ctx + ((long)b * NSEQ + n0 + j * 16) * C + (long)(mt * 16) * NH + h,
                NH, C, c[j]);
}

// ---------------------------------------------------------------------------
// Kernel 6: out[b,n,e] = sum_c ctxFlat[b,n,c] * Wo[e,c]; A feeds 4 N-tiles.
// ---------------------------------------------------------------------------
__global__ __launch_bounds__(256)
void out_gemm_kernel(const _Float16* __restrict__ ctx,
                     const _Float16* __restrict__ Wof,
                     float* __restrict__ out, int C) {
  const int cb = C / 64;   // blocks of 4 e-tiles
  const int wt = blockIdx.x * WPB + (threadIdx.x >> 5);
  const int total = BQ * (NSEQ / 16) * cb;
  if (wt >= total) return;
  const int nb = wt % cb;
  const int mt = (wt / cb) % (NSEQ / 16);
  const int b  = wt / (cb * (NSEQ / 16));
  const int e0 = nb * 64;

  const _Float16* A = ctx + ((long)b * NSEQ + mt * 16) * C;  // sm=C, sk=1
  v8f c[4] = {};
  for (int k0 = 0; k0 < C; k0 += 32) {
    if (k0 + 64 < C) __builtin_prefetch(A + k0 + 64, 0, 1);
    const v16h a = load_a_f16(A, C, 1, k0);
#pragma unroll
    for (int j = 0; j < 4; ++j) {
      // elem(k,n) = Wof[(e0+j*16+n)*C + k]  (sk=1: contiguous per lane)
      const v16h bb = load_b_f16(Wof + (long)(e0 + j * 16) * C, 1, C, k0);
      c[j] = wmma_f16(a, bb, c[j]);
    }
  }
  float* o = out + ((long)b * NSEQ + mt * 16) * C + e0;
#pragma unroll
  for (int j = 0; j < 4; ++j)
    store_d_f32(o + j * 16, C, 1, c[j], 1.f);
}

// ---------------------------------------------------------------------------
extern "C" void kernel_launch(void* const* d_in, const int* in_sizes, int n_in,
                              void* d_out, int out_size, void* d_ws, size_t ws_size,
                              hipStream_t stream) {
  (void)in_sizes; (void)n_in; (void)out_size; (void)ws_size;
  const float* emb[4] = {(const float*)d_in[0], (const float*)d_in[1],
                         (const float*)d_in[2], (const float*)d_in[3]};
  const float* emb_all = (const float*)d_in[4];
  const float* Wq[4] = {(const float*)d_in[5], (const float*)d_in[6],
                        (const float*)d_in[7], (const float*)d_in[8]};
  const float* Wk = (const float*)d_in[9];
  const float* Wv = (const float*)d_in[10];
  const float* Wo[4] = {(const float*)d_in[11], (const float*)d_in[12],
                        (const float*)d_in[13], (const float*)d_in[14]};
  float* out = (float*)d_out;

  // Workspace (f16 unless noted):
  const long HE = (long)BQ * NH * NSEQ * KVP;   // heads / Vh: 16,777,216
  const long TE = (long)BQ * NH * KVC * NSEQ;   // KhT / QhT:  15,728,640
  const long PE = (long)BQ * NH * KVC * KVP;    // probs:       3,932,160
  const long CE = (long)BQ * NSEQ * 960;        // ctx:        15,728,640
  _Float16* heads = (_Float16*)d_ws;
  _Float16* KhT = heads + HE;
  _Float16* QhT = KhT + TE;
  _Float16* Vh  = QhT + TE;
  _Float16* probs = Vh + HE;
  _Float16* ctx = probs + PE;
  _Float16* Wkf = ctx + CE;                     // [240][256]
  _Float16* Wvf = Wkf + KVC * KVP;
  _Float16* Wqf = Wvf + KVC * KVP;              // 2048+4096+16384+65536
  _Float16* Wof = Wqf + (2048 + 4096 + 16384 + 65536);  // 4096+...+262144
  float* scores = (float*)(Wof + (4096 + 16384 + 65536 + 262144));

  const int  dA[4]    = {16, 32, 64, 128};
  const int  dpA[4]   = {32, 32, 64, 128};      // K padded to >=32
  const int  qoffA[4] = {0, 16, 48, 112};
  const int  CA[4]    = {64, 128, 256, 512};
  const long wqoff[4] = {0, 2048, 6144, 22528};
  const long wooff[4] = {0, 4096, 20480, 86016};
  const long ctxoff[4] = {0, (long)BQ * NSEQ * 64, (long)BQ * NSEQ * 192,
                          (long)BQ * NSEQ * 448};
  const long outoff[4] = {0, 1048576, 3145728, 7340032};

  auto nblk = [](long x) { return (int)((x + 255) / 256); };
  auto wblk = [](long waves) { return (int)((waves + WPB - 1) / WPB); };

  // --- prep: regroup + convert weights + zero Vh pads ---
  regroup_kernel<<<nblk(HE), 256, 0, stream>>>(emb_all, heads);
  wkv_prep_kernel<<<nblk(KVC * KVP), 256, 0, stream>>>(Wk, Wv, Wkf, Wvf);
  for (int i = 0; i < 4; ++i)
    wq_prep_kernel<<<nblk((long)NH * dA[i] * dpA[i]), 256, 0, stream>>>(
        Wq[i], Wqf + wqoff[i], dA[i], dpA[i]);
  for (int i = 0; i < 4; ++i)
    wo_prep_kernel<<<nblk((long)CA[i] * CA[i]), 256, 0, stream>>>(
        Wo[i], Wof + wooff[i], CA[i] * CA[i]);
  padv_kernel<<<256, 256, 0, stream>>>(Vh);

  // --- pipeline ---
  kv_gemm_kernel<<<wblk((long)BQ * NH * 64 * 5), 256, 0, stream>>>(
      heads, Wkf, Wvf, KhT, Vh);

  for (int i = 0; i < 4; ++i)
    q_gemm_kernel<<<wblk((long)BQ * NH * 64 * (dA[i] / 16)), 256, 0, stream>>>(
        emb[i], Wqf + wqoff[i], QhT, dA[i], dpA[i], qoffA[i], CA[i]);

  for (int i = 0; i < 4; ++i)
    scores_gemm_kernel<<<wblk((long)BQ * NH * (dA[i] / 16) * 5), 256, 0, stream>>>(
        QhT, KhT, scores, dA[i], qoffA[i]);

  for (int i = 0; i < 4; ++i)
    norm_softmax_kernel<<<BQ * NH, 256, 0, stream>>>(scores, probs, dA[i], qoffA[i]);

  for (int i = 0; i < 4; ++i)
    ctx_gemm_kernel<<<wblk((long)BQ * NH * (dA[i] / 16) * 16), 256, 0, stream>>>(
        probs, Vh, ctx + ctxoff[i], dA[i], qoffA[i], CA[i]);

  for (int i = 0; i < 4; ++i)
    out_gemm_kernel<<<wblk((long)BQ * 64 * (CA[i] / 64)), 256, 0, stream>>>(
        ctx + ctxoff[i], Wof + wooff[i], out + outoff[i], CA[i]);
}
